// graphModel_34780645163111
// MI455X (gfx1250) — compile-verified, matrix operand order checked by reading, not run on hardware
//
#include <hip/hip_runtime.h>
#include <hip/hip_bf16.h>

// ---------------- types ----------------
typedef __attribute__((ext_vector_type(8)))  float  v8f;
typedef __attribute__((ext_vector_type(16))) __bf16 v16bf;
typedef __attribute__((ext_vector_type(8)))  __bf16 v8bf;

union FragBF {
    v16bf v;
    v8bf  h[2];
};

// ---------------- constants ----------------
#define NEG_SLOPE 0.2f
#define EPS_F 1e-16f

#define N_NODES 8192
#define N_EDGES 65536
#define D_IN    768
#define D_HID   768
#define H1      8
#define HC1     (H1 * D_HID)   // 6144

// GEMM tiling
#define BM 128
#define BN 128
#define BK 32
#define LDA 40   // padded LDS stride (elements) for A rows
#define LDB 40   // padded LDS stride (elements) for B cols (transposed)

// ---------------- helpers ----------------
__device__ __forceinline__ unsigned short f2bf(float f) {
    unsigned u = __float_as_uint(f);
    unsigned r = ((u >> 16) & 1u) + 0x7FFFu;   // round-to-nearest-even
    return (unsigned short)((u + r) >> 16);
}
__device__ __forceinline__ float bf2f(unsigned short b) {
    return __uint_as_float(((unsigned)b) << 16);
}
__device__ __forceinline__ void atomicMaxFloat(float* addr, float val) {
    if (val >= 0.0f) atomicMax((int*)addr, __float_as_int(val));
    else             atomicMin((unsigned int*)addr, __float_as_uint(val));
}

// ---------------- kernels ----------------

__global__ __launch_bounds__(256)
void k_cvt_bf16(const float* __restrict__ in, unsigned short* __restrict__ out, long n) {
    long i = (long)blockIdx.x * blockDim.x + threadIdx.x;
    long stride = (long)gridDim.x * blockDim.x;
    for (; i < n; i += stride) out[i] = f2bf(in[i]);
}

__global__ __launch_bounds__(256)
void k_fill_f32(float* __restrict__ p, float v, long n) {
    long i = (long)blockIdx.x * blockDim.x + threadIdx.x;
    long stride = (long)gridDim.x * blockDim.x;
    for (; i < n; i += stride) p[i] = v;
}

// d_out[n*C + c] = b[c]
__global__ __launch_bounds__(256)
void k_init_bias(float* __restrict__ out, const float* __restrict__ b, int C, long n) {
    long i = (long)blockIdx.x * blockDim.x + threadIdx.x;
    long stride = (long)gridDim.x * blockDim.x;
    for (; i < n; i += stride) out[i] = b[i % C];
}

// C[M,N] (bf16) = A[M,K] (bf16) * B[K,N] (bf16), fp32 accumulate via WMMA.
// Double-buffered LDS; global loads for tile k+1 overlap WMMA on tile k.
__global__ __launch_bounds__(256)
void k_gemm_bf16_wmma(const unsigned short* __restrict__ A,
                      const unsigned short* __restrict__ B,
                      unsigned short* __restrict__ C,
                      int M, int N, int K) {
    __shared__ unsigned short sA[2][BM * LDA];
    __shared__ unsigned short sB[2][BN * LDB];

    const int tid    = threadIdx.x;
    const int wid    = tid >> 5;
    const int lane   = tid & 31;
    const int laneLo = lane & 15;
    const int laneHi = lane >> 4;

    const int blockRow = blockIdx.y * BM;
    const int blockCol = blockIdx.x * BN;

    // 8 waves arranged 2 (M) x 4 (N): each wave owns 64 rows x 32 cols
    const int waveM = (wid >> 2) * 64;
    const int waveN = (wid & 3) * 32;

    // ---- staging coordinates (loop-invariant) ----
    // A tile 128x32: thread handles rows {rA, rA+64}, 8 cols at cqA
    const int rA  = tid >> 2;            // 0..63
    const int cqA = (tid & 3) * 8;       // 0,8,16,24
    // B tile 32x128: thread handles K rows {kkB, kkB+16}, 8 cols at nnB
    const int kkB = tid >> 4;            // 0..15
    const int nnB = (tid & 15) * 8;      // 0..120

    // global pointers, advanced by BK (A) / BK*N (B) per K-tile
    const unsigned short* aP = A + (long)(blockRow + rA) * K + cqA;
    const unsigned short* bP = B + (long)kkB * N + blockCol + nnB;
    const long aChunkStride = (long)64 * K;   // second A chunk: +64 rows
    const long bChunkStride = (long)16 * N;   // second B chunk: +16 K-rows
    const long bStep = (long)BK * N;

    // LDS store offsets (elements)
    const int sAoff0 = rA * LDA + cqA;
    const int sAoff1 = (rA + 64) * LDA + cqA;
    const int sBoffBase = nnB * LDB + kkB;    // +j*LDB per element, +16 for chunk 1

    // fragment read offsets (elements)
    int aOff[4], bOff[2];
    #pragma unroll
    for (int mi = 0; mi < 4; ++mi)
        aOff[mi] = (waveM + mi * 16 + laneLo) * LDA + laneHi * 8;
    #pragma unroll
    for (int ni = 0; ni < 2; ++ni)
        bOff[ni] = (waveN + ni * 16 + laneLo) * LDB + laneHi * 8;

    v8f acc[4][2] = {};
    uint4 ra0, ra1, rb0, rb1;

    const int kTiles = K / BK;

    // prologue: load tile 0 -> regs -> lds[0]
    ra0 = *(const uint4*)(aP);
    ra1 = *(const uint4*)(aP + aChunkStride);
    rb0 = *(const uint4*)(bP);
    rb1 = *(const uint4*)(bP + bChunkStride);
    aP += BK;
    bP += bStep;
    {
        *(uint4*)(&sA[0][sAoff0]) = ra0;
        *(uint4*)(&sA[0][sAoff1]) = ra1;
        const unsigned short* p0 = (const unsigned short*)&rb0;
        const unsigned short* p1 = (const unsigned short*)&rb1;
        #pragma unroll
        for (int j = 0; j < 8; ++j) {
            sB[0][sBoffBase + j * LDB]      = p0[j];
            sB[0][sBoffBase + j * LDB + 16] = p1[j];
        }
    }
    __syncthreads();

    for (int kt = 0; kt < kTiles; ++kt) {
        const int buf = kt & 1;
        const bool more = (kt + 1 < kTiles);

        // issue global loads for tile kt+1 (latency hidden under WMMA)
        if (more) {
            ra0 = *(const uint4*)(aP);
            ra1 = *(const uint4*)(aP + aChunkStride);
            rb0 = *(const uint4*)(bP);
            rb1 = *(const uint4*)(bP + bChunkStride);
            aP += BK;
            bP += bStep;
            if (kt + 2 < kTiles) {
                __builtin_prefetch(aP, 0, 1);
                __builtin_prefetch(bP, 0, 1);
            }
        }

        // ---- compute: 8 WMMA from lds[buf] ----
        const unsigned short* __restrict__ cA = sA[buf];
        const unsigned short* __restrict__ cB = sB[buf];
        FragBF bf0, bf1;
        bf0.h[0] = *(const v8bf*)(cB + bOff[0]);
        bf0.h[1] = *(const v8bf*)(cB + bOff[0] + 16);
        bf1.h[0] = *(const v8bf*)(cB + bOff[1]);
        bf1.h[1] = *(const v8bf*)(cB + bOff[1] + 16);
        #pragma unroll
        for (int mi = 0; mi < 4; ++mi) {
            FragBF a;
            a.h[0] = *(const v8bf*)(cA + aOff[mi]);
            a.h[1] = *(const v8bf*)(cA + aOff[mi] + 16);
            acc[mi][0] = __builtin_amdgcn_wmma_f32_16x16x32_bf16(
                false, a.v, false, bf0.v, (short)0, acc[mi][0], false, false);
            acc[mi][1] = __builtin_amdgcn_wmma_f32_16x16x32_bf16(
                false, a.v, false, bf1.v, (short)0, acc[mi][1], false, false);
        }

        // ---- drain loads into the other buffer, one barrier per iteration ----
        if (more) {
            const int nbuf = buf ^ 1;
            *(uint4*)(&sA[nbuf][sAoff0]) = ra0;
            *(uint4*)(&sA[nbuf][sAoff1]) = ra1;
            const unsigned short* p0 = (const unsigned short*)&rb0;
            const unsigned short* p1 = (const unsigned short*)&rb1;
            #pragma unroll
            for (int j = 0; j < 8; ++j) {
                sB[nbuf][sBoffBase + j * LDB]      = p0[j];
                sB[nbuf][sBoffBase + j * LDB + 16] = p1[j];
            }
            __syncthreads();
        }
    }

    // ---- store C as bf16; 16x16 f32 D layout: VGPR r -> M = r + laneHi*8, N = laneLo ----
    #pragma unroll
    for (int mi = 0; mi < 4; ++mi) {
        #pragma unroll
        for (int ni = 0; ni < 2; ++ni) {
            const int col = blockCol + waveN + ni * 16 + laneLo;
            #pragma unroll
            for (int r = 0; r < 8; ++r) {
                const int row = blockRow + waveM + mi * 16 + laneHi * 8 + r;
                C[(long)row * N + col] = f2bf(acc[mi][ni][r]);
            }
        }
    }
}

// alpha_s[n,h] = sum_c h[n,h,c]*a_s[h,c]; same for a_d. One wave per (n,h).
__global__ __launch_bounds__(256)
void k_alpha_proj(const unsigned short* __restrict__ h,
                  const float* __restrict__ a_s, const float* __restrict__ a_d,
                  float* __restrict__ out_s, float* __restrict__ out_d,
                  int nNodes, int H, int C) {
    const int gwarp = (int)((blockIdx.x * blockDim.x + threadIdx.x) >> 5);
    const int lane  = threadIdx.x & 31;
    if (gwarp >= nNodes * H) return;
    const int n  = gwarp / H;
    const int hh = gwarp % H;
    const unsigned short* hp = h + (long)n * H * C + (long)hh * C;
    const float* asp = a_s + hh * C;
    const float* adp = a_d + hh * C;
    float ss = 0.0f, sd = 0.0f;
    for (int c = lane; c < C; c += 32) {
        float v = bf2f(hp[c]);
        ss += v * asp[c];
        sd += v * adp[c];
    }
    #pragma unroll
    for (int off = 16; off > 0; off >>= 1) {
        ss += __shfl_xor(ss, off, 32);
        sd += __shfl_xor(sd, off, 32);
    }
    if (lane == 0) { out_s[gwarp] = ss; out_d[gwarp] = sd; }
}

// e = leaky_relu(as[src]+ad[dst]); segment max over dst via atomic
__global__ __launch_bounds__(256)
void k_edge_scores(const int* __restrict__ src, const int* __restrict__ dst,
                   const float* __restrict__ as, const float* __restrict__ ad,
                   float* __restrict__ e, float* __restrict__ m, int E, int H) {
    int i = blockIdx.x * blockDim.x + threadIdx.x;
    if (i >= E * H) return;
    int ed = i / H, hh = i % H;
    int s = src[ed], d = dst[ed];
    float v = as[s * H + hh] + ad[d * H + hh];
    v = (v > 0.0f) ? v : v * NEG_SLOPE;
    e[i] = v;
    atomicMaxFloat(&m[d * H + hh], v);
}

// e = exp(e - m[dst]); segment sum into den
__global__ __launch_bounds__(256)
void k_edge_exp(const int* __restrict__ dst, float* __restrict__ e,
                const float* __restrict__ m, float* __restrict__ den, int E, int H) {
    int i = blockIdx.x * blockDim.x + threadIdx.x;
    if (i >= E * H) return;
    int ed = i / H, hh = i % H;
    int d = dst[ed];
    float v = __expf(e[i] - m[d * H + hh]);
    e[i] = v;
    atomicAdd(&den[d * H + hh], v);
}

// per edge: out[dst] += h[src] * (e/(den[dst]+eps)), one block per edge
__global__ __launch_bounds__(256)
void k_aggregate(const unsigned short* __restrict__ h,
                 const float* __restrict__ e, const float* __restrict__ den,
                 const int* __restrict__ src, const int* __restrict__ dst,
                 float* __restrict__ out, int H, int C) {
    __shared__ float alphaSh[8];
    const int ed = blockIdx.x;
    const int s = src[ed], d = dst[ed];
    const int HC = H * C;
    if (threadIdx.x < H)
        alphaSh[threadIdx.x] =
            e[(long)ed * H + threadIdx.x] / (den[(long)d * H + threadIdx.x] + EPS_F);
    __syncthreads();
    for (int idx = threadIdx.x; idx < HC; idx += blockDim.x) {
        const int hh = idx / C;
        const float v = bf2f(h[(long)s * HC + idx]) * alphaSh[hh];
        atomicAdd(&out[(long)d * HC + idx], v);
    }
}

// out_bf16 = elu(agg + b)
__global__ __launch_bounds__(256)
void k_elu_bias_cvt(const float* __restrict__ agg, const float* __restrict__ b,
                    unsigned short* __restrict__ out, int HC, long n) {
    long i = (long)blockIdx.x * blockDim.x + threadIdx.x;
    long stride = (long)gridDim.x * blockDim.x;
    for (; i < n; i += stride) {
        float v = agg[i] + b[i % HC];
        v = (v > 0.0f) ? v : (__expf(v) - 1.0f);
        out[i] = f2bf(v);
    }
}

// ---------------- host launcher ----------------
extern "C" void kernel_launch(void* const* d_in, const int* in_sizes, int n_in,
                              void* d_out, int out_size, void* d_ws, size_t ws_size,
                              hipStream_t stream) {
    (void)in_sizes; (void)n_in; (void)out_size; (void)ws_size;

    const float* x      = (const float*)d_in[0];
    const float* W1     = (const float*)d_in[1];
    const float* a_src1 = (const float*)d_in[2];
    const float* a_dst1 = (const float*)d_in[3];
    const float* b1     = (const float*)d_in[4];
    const float* W2     = (const float*)d_in[5];
    const float* a_src2 = (const float*)d_in[6];
    const float* a_dst2 = (const float*)d_in[7];
    const float* b2     = (const float*)d_in[8];
    const int*   edges  = (const int*)d_in[9];
    const int* src = edges;
    const int* dst = edges + N_EDGES;
    float* out = (float*)d_out;

    // carve workspace
    char* ws = (char*)d_ws;
    size_t off = 0;
    auto carve = [&](size_t bytes) -> char* {
        char* p = ws + off;
        off = (off + bytes + 255) & ~(size_t)255;
        return p;
    };
    unsigned short* xb   = (unsigned short*)carve((size_t)N_NODES * D_IN * 2);
    unsigned short* w1b  = (unsigned short*)carve((size_t)D_IN * HC1 * 2);
    unsigned short* w2b  = (unsigned short*)carve((size_t)HC1 * D_HID * 2);
    unsigned short* h1b  = (unsigned short*)carve((size_t)N_NODES * HC1 * 2); // reused for elu output
    unsigned short* h2b  = (unsigned short*)carve((size_t)N_NODES * D_HID * 2);
    float* as1  = (float*)carve((size_t)N_NODES * H1 * 4);
    float* ad1  = (float*)carve((size_t)N_NODES * H1 * 4);
    float* m1   = (float*)carve((size_t)N_NODES * H1 * 4);
    float* den1 = (float*)carve((size_t)N_NODES * H1 * 4);
    float* e1   = (float*)carve((size_t)N_EDGES * H1 * 4);
    float* agg1 = (float*)carve((size_t)N_NODES * HC1 * 4);
    float* as2  = (float*)carve((size_t)N_NODES * 4);
    float* ad2  = (float*)carve((size_t)N_NODES * 4);
    float* m2   = (float*)carve((size_t)N_NODES * 4);
    float* den2 = (float*)carve((size_t)N_NODES * 4);
    float* e2   = (float*)carve((size_t)N_EDGES * 4);

    const float NEG_INF = -__builtin_huge_valf();
    auto blocks = [](long n) { return (unsigned)((n + 255) / 256); };

    // convert inputs to bf16
    k_cvt_bf16<<<2048, 256, 0, stream>>>(x, xb, (long)N_NODES * D_IN);
    k_cvt_bf16<<<2048, 256, 0, stream>>>(W1, w1b, (long)D_IN * HC1);
    k_cvt_bf16<<<2048, 256, 0, stream>>>(W2, w2b, (long)HC1 * D_HID);

    // ---- layer 1 ----
    // h1 = x @ W1  [8192 x 6144]
    k_gemm_bf16_wmma<<<dim3(HC1 / BN, N_NODES / BM), 256, 0, stream>>>(
        xb, w1b, h1b, N_NODES, HC1, D_IN);
    // attention score projections
    k_alpha_proj<<<blocks((long)N_NODES * H1 * 32), 256, 0, stream>>>(
        h1b, a_src1, a_dst1, as1, ad1, N_NODES, H1, D_HID);
    // init softmax state + accumulator
    k_fill_f32<<<blocks((long)N_NODES * H1), 256, 0, stream>>>(m1, NEG_INF, (long)N_NODES * H1);
    k_fill_f32<<<blocks((long)N_NODES * H1), 256, 0, stream>>>(den1, 0.0f, (long)N_NODES * H1);
    k_fill_f32<<<8192, 256, 0, stream>>>(agg1, 0.0f, (long)N_NODES * HC1);
    // edge softmax
    k_edge_scores<<<blocks((long)N_EDGES * H1), 256, 0, stream>>>(src, dst, as1, ad1, e1, m1, N_EDGES, H1);
    k_edge_exp<<<blocks((long)N_EDGES * H1), 256, 0, stream>>>(dst, e1, m1, den1, N_EDGES, H1);
    // message aggregation
    k_aggregate<<<N_EDGES, 256, 0, stream>>>(h1b, e1, den1, src, dst, agg1, H1, D_HID);
    // elu(agg + b1) -> bf16 (overwrites h1b)
    k_elu_bias_cvt<<<8192, 256, 0, stream>>>(agg1, b1, h1b, HC1, (long)N_NODES * HC1);

    // ---- layer 2 ----
    // h2 = h1act @ W2  [8192 x 768]
    k_gemm_bf16_wmma<<<dim3(D_HID / BN, N_NODES / BM), 256, 0, stream>>>(
        h1b, w2b, h2b, N_NODES, D_HID, HC1);
    k_alpha_proj<<<blocks((long)N_NODES * 32), 256, 0, stream>>>(
        h2b, a_src2, a_dst2, as2, ad2, N_NODES, 1, D_HID);
    k_fill_f32<<<blocks((long)N_NODES), 256, 0, stream>>>(m2, NEG_INF, (long)N_NODES);
    k_fill_f32<<<blocks((long)N_NODES), 256, 0, stream>>>(den2, 0.0f, (long)N_NODES);
    // d_out = broadcast(b2); aggregation accumulates on top
    k_init_bias<<<blocks((long)N_NODES * D_HID), 256, 0, stream>>>(out, b2, D_HID, (long)N_NODES * D_HID);
    k_edge_scores<<<blocks((long)N_EDGES), 256, 0, stream>>>(src, dst, as2, ad2, e2, m2, N_EDGES, 1);
    k_edge_exp<<<blocks((long)N_EDGES), 256, 0, stream>>>(dst, e2, m2, den2, N_EDGES, 1);
    k_aggregate<<<N_EDGES, 256, 0, stream>>>(h2b, e2, den2, src, dst, out, 1, D_HID);
}
